// DynamicClusterModel_26886495273499
// MI455X (gfx1250) — matrix operand order repaired and checked by reading, not compiled, not run on hardware
//
#include <hip/hip_runtime.h>

typedef __attribute__((ext_vector_type(16))) _Float16 v16h;
typedef __attribute__((ext_vector_type(8)))  _Float16 v8h;
typedef __attribute__((ext_vector_type(8)))  float    v8f;
typedef __attribute__((ext_vector_type(4)))  float    v4f;
typedef __attribute__((ext_vector_type(4)))  int      v4i;

constexpr int CDIM = 32;     // feature dim
constexpr int KCL  = 4096;   // number of clusters

// ---------------------------------------------------------------- k_zero ----
__global__ void k_zero(float* __restrict__ p, int n) {
    int i = blockIdx.x * blockDim.x + threadIdx.x;
    if (i < n) p[i] = 0.0f;
}

// -------------------------------------------------------- k_dense_accum ----
// Each wave handles one 16-row tile per loop iteration:
//   x = relu(relu(feats @ Wd1 + bd1) @ Wd2 + bd2)  via v_wmma_f32_16x16x32_f16
// then atomically accumulates per-cluster sums + counts (L2-resident array).
__global__ void k_dense_accum(const float* __restrict__ feats,
                              const int*   __restrict__ ids,
                              const float* __restrict__ Wd1, const float* __restrict__ bd1,
                              const float* __restrict__ Wd2, const float* __restrict__ bd2,
                              float* __restrict__ sums, float* __restrict__ cnts,
                              int ntiles)
{
    __shared__ _Float16 xtile[8][16 * CDIM];   // per-wave staging (8 waves @ 1 KB)

    const int lane = threadIdx.x & 31;
    const int wave = threadIdx.x >> 5;
    const int half = lane >> 4;     // 0: lanes 0-15, 1: lanes 16-31
    const int lr   = lane & 15;

    // --- preload B fragments (32x16 f16, ISA B layout) for Wd1/Wd2 col-blocks
    // lane element j = W[k*32 + (cb*16 + lr)], k = j<8 ? half*8+j : 16+half*8+(j-8)
    v16h b1[2], b2[2];
    #pragma unroll
    for (int cb = 0; cb < 2; ++cb) {
        const int col = cb * 16 + lr;
        #pragma unroll
        for (int j = 0; j < 16; ++j) {
            const int k = (j < 8) ? (half * 8 + j) : (16 + half * 8 + (j - 8));
            b1[cb][j] = (_Float16)Wd1[k * CDIM + col];
            b2[cb][j] = (_Float16)Wd2[k * CDIM + col];
        }
    }
    const float bias1[2] = { bd1[lr], bd1[16 + lr] };
    const float bias2[2] = { bd2[lr], bd2[16 + lr] };

    _Float16* tile = &xtile[wave][0];

    const int gwave  = blockIdx.x * (blockDim.x >> 5) + wave;
    const int nwaves = gridDim.x * (blockDim.x >> 5);

    for (int t = gwave; t < ntiles; t += nwaves) {
        const size_t R = (size_t)t * 16;

        // Prefetch next tile's row for this lane (speculative; dropped if OOB).
        __builtin_prefetch(feats + ((size_t)(t + nwaves) * 16 + lr) * CDIM, 0, 1);

        // --- A fragment (16x32 f16): row = R+lr, lane<16 holds K 0-7/16-23,
        //     lane>=16 holds K 8-15/24-31 (two aligned 32B non-temporal loads:
        //     feats is streamed once, keep it out of the hot L2 set)
        const float* rowp = feats + (R + lr) * CDIM + half * 8;
        v8f flo = __builtin_nontemporal_load((const v8f*)(rowp));
        v8f fhi = __builtin_nontemporal_load((const v8f*)(rowp + 16));
        v8h hlo = __builtin_convertvector(flo, v8h);
        v8h hhi = __builtin_convertvector(fhi, v8h);
        v16h a  = __builtin_shufflevector(hlo, hhi,
                    0,1,2,3,4,5,6,7,8,9,10,11,12,13,14,15);

        // --- layer 1: D(16x16) x2 column halves
        v8f d0 = {}, d1 = {};
        d0 = __builtin_amdgcn_wmma_f32_16x16x32_f16(false, a, false, b1[0],
                                                    (short)0, d0, false, false);
        d1 = __builtin_amdgcn_wmma_f32_16x16x32_f16(false, a, false, b1[1],
                                                    (short)0, d1, false, false);

        // bias + relu -> stage f16 tile in LDS (row-major 16x32)
        #pragma unroll
        for (int v = 0; v < 8; ++v) {
            const int row = half * 8 + v;                   // D layout: M = half*8+v
            float x0 = d0[v] + bias1[0]; x0 = x0 > 0.0f ? x0 : 0.0f;
            float x1 = d1[v] + bias1[1]; x1 = x1 > 0.0f ? x1 : 0.0f;
            tile[row * CDIM + lr]      = (_Float16)x0;       // col = lr
            tile[row * CDIM + 16 + lr] = (_Float16)x1;       // col = 16+lr
        }
        asm volatile("s_wait_dscnt 0" ::: "memory");        // wave-local producer/consumer

        // --- reload tile as A fragment for layer 2 (two aligned 16B LDS loads)
        const _Float16* trow = tile + lr * CDIM + half * 8;
        v8h tlo = *(const v8h*)(trow);
        v8h thi = *(const v8h*)(trow + 16);
        v16h a2 = __builtin_shufflevector(tlo, thi,
                    0,1,2,3,4,5,6,7,8,9,10,11,12,13,14,15);

        v8f e0 = {}, e1 = {};
        e0 = __builtin_amdgcn_wmma_f32_16x16x32_f16(false, a2, false, b2[0],
                                                    (short)0, e0, false, false);
        e1 = __builtin_amdgcn_wmma_f32_16x16x32_f16(false, a2, false, b2[1],
                                                    (short)0, e1, false, false);

        // --- cluster ids for this lane's 8 rows: two broadcast b128 loads
        const v4i* idp = (const v4i*)(ids + R + half * 8);
        const v4i idA = idp[0];
        const v4i idB = idp[1];
        const int idv[8] = { idA.x, idA.y, idA.z, idA.w,
                             idB.x, idB.y, idB.z, idB.w };

        // --- bias + relu, atomic segment-sum into L2-resident cluster array
        #pragma unroll
        for (int v = 0; v < 8; ++v) {
            const int id = idv[v];
            float y0 = e0[v] + bias2[0]; y0 = y0 > 0.0f ? y0 : 0.0f;
            float y1 = e1[v] + bias2[1]; y1 = y1 > 0.0f ? y1 : 0.0f;
            float* base = sums + (size_t)id * CDIM;
            atomicAdd(base + lr,      y0);
            atomicAdd(base + 16 + lr, y1);
            if (lr == 0) atomicAdd(cnts + id, 1.0f);
        }
    }
}

// -------------------------------------------------------- k_cluster_mlp ----
// One thread per cluster: mean -> 3x (32x32 relu) -> 32x2 head -> gumbel argmax.
__global__ void k_cluster_mlp(const float* __restrict__ sums,
                              const float* __restrict__ cnts,
                              const float* __restrict__ gumbel,
                              const float* __restrict__ Wc1, const float* __restrict__ bc1,
                              const float* __restrict__ Wc2, const float* __restrict__ bc2,
                              const float* __restrict__ Wc3, const float* __restrict__ bc3,
                              const float* __restrict__ Wc4, const float* __restrict__ bc4,
                              float* __restrict__ flags)
{
    __shared__ float sW1[CDIM * CDIM], sW2[CDIM * CDIM], sW3[CDIM * CDIM];
    __shared__ float sW4[CDIM * 2];
    __shared__ float sb1[CDIM], sb2[CDIM], sb3[CDIM], sb4[2];

    for (int i = threadIdx.x; i < CDIM * CDIM; i += blockDim.x) {
        sW1[i] = Wc1[i]; sW2[i] = Wc2[i]; sW3[i] = Wc3[i];
    }
    for (int i = threadIdx.x; i < CDIM * 2; i += blockDim.x) sW4[i] = Wc4[i];
    if (threadIdx.x < CDIM) {
        sb1[threadIdx.x] = bc1[threadIdx.x];
        sb2[threadIdx.x] = bc2[threadIdx.x];
        sb3[threadIdx.x] = bc3[threadIdx.x];
    }
    if (threadIdx.x < 2) sb4[threadIdx.x] = bc4[threadIdx.x];
    __syncthreads();

    const int k = blockIdx.x * blockDim.x + threadIdx.x;
    if (k >= KCL) return;

    const float inv = 1.0f / fmaxf(cnts[k], 1.0f);
    float h0[CDIM], h1[CDIM];
    #pragma unroll
    for (int c = 0; c < CDIM; ++c) h0[c] = sums[(size_t)k * CDIM + c] * inv;

    #pragma unroll
    for (int o = 0; o < CDIM; ++o) {
        float acc = sb1[o];
        #pragma unroll
        for (int i = 0; i < CDIM; ++i) acc += h0[i] * sW1[i * CDIM + o];
        h1[o] = fmaxf(acc, 0.0f);
    }
    #pragma unroll
    for (int o = 0; o < CDIM; ++o) {
        float acc = sb2[o];
        #pragma unroll
        for (int i = 0; i < CDIM; ++i) acc += h1[i] * sW2[i * CDIM + o];
        h0[o] = fmaxf(acc, 0.0f);
    }
    #pragma unroll
    for (int o = 0; o < CDIM; ++o) {
        float acc = sb3[o];
        #pragma unroll
        for (int i = 0; i < CDIM; ++i) acc += h0[i] * sW3[i * CDIM + o];
        h1[o] = fmaxf(acc, 0.0f);
    }

    float l0 = sb4[0], l1 = sb4[1];
    #pragma unroll
    for (int i = 0; i < CDIM; ++i) {
        l0 += h1[i] * sW4[i * 2 + 0];
        l1 += h1[i] * sW4[i * 2 + 1];
    }
    l0 += gumbel[k * 2 + 0];
    l1 += gumbel[k * 2 + 1];

    // softmax is monotone; argmax==1 iff l1+g1 strictly greater (ties -> index 0)
    flags[k] = (l1 > l0) ? 1.0f : 0.0f;
}

// ------------------------------------------------------------- k_gather ----
// 4 rows per thread: b128 id loads, non-temporal b128 output stores.
__global__ void k_gather4(const v4i* __restrict__ ids4,
                          const float* __restrict__ flags,
                          v4f* __restrict__ out4, int n4)
{
    int i = blockIdx.x * blockDim.x + threadIdx.x;
    if (i < n4) {
        const v4i id = ids4[i];
        v4f r;
        r.x = flags[id.x];
        r.y = flags[id.y];
        r.z = flags[id.z];
        r.w = flags[id.w];
        __builtin_nontemporal_store(r, out4 + i);
    }
}

// -------------------------------------------------------------- launch -----
extern "C" void kernel_launch(void* const* d_in, const int* in_sizes, int n_in,
                              void* d_out, int out_size, void* d_ws, size_t ws_size,
                              hipStream_t stream)
{
    (void)n_in; (void)out_size; (void)ws_size;

    const float* feats  = (const float*)d_in[0];
    const int*   ids    = (const int*)  d_in[1];
    const float* gumbel = (const float*)d_in[2];
    const float* Wd1 = (const float*)d_in[3];
    const float* bd1 = (const float*)d_in[4];
    const float* Wd2 = (const float*)d_in[5];
    const float* bd2 = (const float*)d_in[6];
    const float* Wc1 = (const float*)d_in[7];
    const float* bc1 = (const float*)d_in[8];
    const float* Wc2 = (const float*)d_in[9];
    const float* bc2 = (const float*)d_in[10];
    const float* Wc3 = (const float*)d_in[11];
    const float* bc3 = (const float*)d_in[12];
    const float* Wc4 = (const float*)d_in[13];
    const float* bc4 = (const float*)d_in[14];
    float* out = (float*)d_out;

    const int n      = in_sizes[0] / CDIM;   // N rows
    const int ntiles = n / 16;               // N = 2,000,000 is a multiple of 16

    float* sums  = (float*)d_ws;             // KCL * 32 floats (512 KB, L2-resident)
    float* cnts  = sums + (size_t)KCL * CDIM;
    float* flags = cnts + KCL;

    const int nzero = KCL * CDIM + KCL;
    k_zero<<<(nzero + 255) / 256, 256, 0, stream>>>(sums, nzero);

    k_dense_accum<<<1024, 256, 0, stream>>>(feats, ids, Wd1, bd1, Wd2, bd2,
                                            sums, cnts, ntiles);

    k_cluster_mlp<<<KCL / 256, 256, 0, stream>>>(sums, cnts, gumbel,
                                                 Wc1, bc1, Wc2, bc2,
                                                 Wc3, bc3, Wc4, bc4, flags);

    const int n4 = n / 4;                    // N is a multiple of 4
    k_gather4<<<(n4 + 255) / 256, 256, 0, stream>>>((const v4i*)ids, flags,
                                                    (v4f*)out, n4);
}
